// EdgeSAGE_43456479101296
// MI455X (gfx1250) — compile-verified, matrix operand order checked by reading, not run on hardware
//
#include <hip/hip_runtime.h>
#include <hip/hip_bf16.h>

// ---------------------------------------------------------------------------
// EdgeSAGE on gfx1250: edge-MLP -> 3x (scatter-mean + dual-GEMM via WMMA bf16)
// Dense parts use v_wmma_f32_16x16x32_bf16 (f32 accumulate).
// Scatter uses hardware global_atomic_add_f32 (unsafeAtomicAdd).
// ---------------------------------------------------------------------------

typedef __attribute__((ext_vector_type(16))) __bf16 v16bf;
typedef __attribute__((ext_vector_type(8)))  float  v8f;

#define NN    100000
#define EE    1600000
#define FIN   128
#define HID   128

static __device__ __forceinline__ __bf16 f2bf(float f) {
    union { float f; unsigned u; } x; x.f = f;
    unsigned r = (x.u + 0x7FFFu + ((x.u >> 16) & 1u)) >> 16;   // RNE
    union { unsigned short s; __bf16 b; } y; y.s = (unsigned short)r;
    return y.b;
}

// ---------------- edge MLP: ew = sigmoid(relu(ea@W1+b1)@W2+b2) --------------
__global__ void edge_mlp_kernel(const float* __restrict__ ea,
                                const float* __restrict__ w1, const float* __restrict__ b1,
                                const float* __restrict__ w2, const float* __restrict__ b2,
                                float* __restrict__ ew, int E_) {
    __shared__ float sw1[8 * 128];
    __shared__ float sb1[128];
    __shared__ float sw2[128];
    for (int i = threadIdx.x; i < 1024; i += blockDim.x) sw1[i] = w1[i];
    for (int i = threadIdx.x; i < 128;  i += blockDim.x) { sb1[i] = b1[i]; sw2[i] = w2[i]; }
    __syncthreads();
    int e = blockIdx.x * blockDim.x + threadIdx.x;
    if (e >= E_) return;
    const float4* p = (const float4*)(ea + (size_t)e * 8);
    float4 v0 = p[0], v1 = p[1];
    float a0=v0.x, a1=v0.y, a2=v0.z, a3=v0.w, a4=v1.x, a5=v1.y, a6=v1.z, a7=v1.w;
    float acc = b2[0];
    #pragma unroll 4
    for (int j = 0; j < 128; ++j) {
        float t = sb1[j];
        t = fmaf(a0, sw1[0*128+j], t); t = fmaf(a1, sw1[1*128+j], t);
        t = fmaf(a2, sw1[2*128+j], t); t = fmaf(a3, sw1[3*128+j], t);
        t = fmaf(a4, sw1[4*128+j], t); t = fmaf(a5, sw1[5*128+j], t);
        t = fmaf(a6, sw1[6*128+j], t); t = fmaf(a7, sw1[7*128+j], t);
        t = fmaxf(t, 0.0f);
        acc = fmaf(t, sw2[j], acc);
    }
    ew[e] = 1.0f / (1.0f + __expf(-acc));
}

// ---------------- degree + reciprocal --------------------------------------
__global__ void deg_kernel(const int* __restrict__ ei, float* __restrict__ deg, int E_) {
    int e = blockIdx.x * blockDim.x + threadIdx.x;
    if (e < E_) unsafeAtomicAdd(&deg[ei[E_ + e]], 1.0f);
}
__global__ void rdeg_kernel(float* __restrict__ deg, int n) {
    int i = blockIdx.x * blockDim.x + threadIdx.x;
    if (i < n) deg[i] = 1.0f / fmaxf(deg[i], 1.0f);
}

// ---------------- scatter: s[dst] += x[src]*ew  (32 lanes/edge) -------------
__global__ void scatter_kernel(const float* __restrict__ x, const int* __restrict__ ei,
                               const float* __restrict__ ew, float* __restrict__ s, int E_) {
    long long g = (long long)blockIdx.x * blockDim.x + threadIdx.x;
    int e = (int)(g >> 5);
    if (e >= E_) return;
    int lane = (int)(g & 31);
    int src = ei[e], dst = ei[E_ + e];
    float w = ew[e];
    float4 v = *(const float4*)(x + (size_t)src * FIN + lane * 4);
    float* o = s + (size_t)dst * FIN + lane * 4;
    unsafeAtomicAdd(o + 0, v.x * w);
    unsafeAtomicAdd(o + 1, v.y * w);
    unsafeAtomicAdd(o + 2, v.z * w);
    unsafeAtomicAdd(o + 3, v.w * w);
}

// ---------------- agg = s*rdeg -> bf16 ; xb = bf16(in) ----------------------
__global__ void aggfin_kernel(const float* __restrict__ s, const float* __restrict__ rdeg,
                              const float* __restrict__ in, __bf16* __restrict__ aggb,
                              __bf16* __restrict__ xb, int total) {
    int i = blockIdx.x * blockDim.x + threadIdx.x;
    if (i >= total) return;
    int node = i >> 7;  // /128
    aggb[i] = f2bf(s[i] * rdeg[node]);
    xb[i]   = f2bf(in[i]);
}

// ---------------- pack weight [128, ncols] into WMMA B-fragment layout ------
// out[((kt*nt_count+nt)*32 + lane)*16 + j] = W[kt*32 + (lane>>4)*16 + j][nt*16 + (lane&15)]
__global__ void pack_w_kernel(const float* __restrict__ w, __bf16* __restrict__ out,
                              int ncols, int nt_count) {
    int idx = blockIdx.x * blockDim.x + threadIdx.x;
    int total = 4 * nt_count * 512;           // KT=4 k-tiles of 32
    if (idx >= total) return;
    int j    = idx & 15;
    int lane = (idx >> 4) & 31;
    int tile = idx >> 9;
    int nt   = tile % nt_count;
    int kt   = tile / nt_count;
    int k = kt * 32 + ((lane >> 4) * 16) + j;
    int n = nt * 16 + (lane & 15);
    float val = (n < ncols) ? w[(size_t)k * ncols + n] : 0.0f;
    out[idx] = f2bf(val);
}

// ---------------- fused dual-GEMM + bias + BN + ReLU ------------------------
// out = agg @ Wl + bias + x @ Wr ; optional BN+ReLU. One wave per 16x16 tile.
__global__ void sage_gemm_kernel(const __bf16* __restrict__ aggb, const __bf16* __restrict__ xb,
                                 const __bf16* __restrict__ wlp,  const __bf16* __restrict__ wrp,
                                 const float* __restrict__ bias,
                                 const float* __restrict__ bng, const float* __restrict__ bnb,
                                 const float* __restrict__ bnm, const float* __restrict__ bnv,
                                 int use_bn, float* __restrict__ out, int out_ld, int out_cols,
                                 int nt_count, int row_tiles) {
    int wave = (blockIdx.x * blockDim.x + threadIdx.x) >> 5;  // wave-uniform
    if (wave >= row_tiles * nt_count) return;                 // uniform exit: EXEC stays all-1s
    int lane  = threadIdx.x & 31;
    int rt    = wave / nt_count;
    int nt    = wave % nt_count;
    int m_lo  = lane & 15;
    int khalf = lane >> 4;                                    // 0 or 1

    const __bf16* arow = aggb + (size_t)(rt * 16 + m_lo) * 128 + khalf * 16;
    const __bf16* xrow = xb   + (size_t)(rt * 16 + m_lo) * 128 + khalf * 16;
    const __bf16* bl   = wlp + ((size_t)nt * 32 + lane) * 16;
    const __bf16* br   = wrp + ((size_t)nt * 32 + lane) * 16;
    size_t kstep = (size_t)nt_count * 512;

    v8f acc = {};
    #pragma unroll
    for (int kt = 0; kt < 4; ++kt) {
        v16bf a  = *(const v16bf*)(arow + kt * 32);
        v16bf b  = *(const v16bf*)(bl + kt * kstep);
        acc = __builtin_amdgcn_wmma_f32_16x16x32_bf16(false, a, false, b, (short)0, acc, false, false);
        v16bf a2 = *(const v16bf*)(xrow + kt * 32);
        v16bf b2 = *(const v16bf*)(br + kt * kstep);
        acc = __builtin_amdgcn_wmma_f32_16x16x32_bf16(false, a2, false, b2, (short)0, acc, false, false);
    }

    int n  = nt * 16 + (lane & 15);
    int np = (n < out_cols) ? n : (out_cols - 1);   // clamp param reads (layer 3)
    float bvv = bias[np];
    float sc = 1.0f, sh = 0.0f;
    if (use_bn) {
        sc = bng[np] * rsqrtf(bnv[np] + 1e-5f);
        sh = bnb[np] - bnm[np] * sc;
    }
    #pragma unroll
    for (int r = 0; r < 8; ++r) {
        int m = r + (khalf ? 8 : 0);
        float h = acc[r] + bvv;
        if (use_bn) h = fmaxf(fmaf(h, sc, sh), 0.0f);
        if (n < out_cols)
            out[(size_t)(rt * 16 + m) * out_ld + n] = h;
    }
}

// ---------------------------------------------------------------------------
extern "C" void kernel_launch(void* const* d_in, const int* in_sizes, int n_in,
                              void* d_out, int out_size, void* d_ws, size_t ws_size,
                              hipStream_t stream) {
    const float* x       = (const float*)d_in[0];
    const float* ea      = (const float*)d_in[1];
    const int*   ei      = (const int*)  d_in[2];
    const float* ew1_w   = (const float*)d_in[3];
    const float* ew1_b   = (const float*)d_in[4];
    const float* ew2_w   = (const float*)d_in[5];
    const float* ew2_b   = (const float*)d_in[6];
    const float* w1_l    = (const float*)d_in[7];
    const float* b1_l    = (const float*)d_in[8];
    const float* w1_r    = (const float*)d_in[9];
    const float* w2_l    = (const float*)d_in[10];
    const float* b2_l    = (const float*)d_in[11];
    const float* w2_r    = (const float*)d_in[12];
    const float* w3_l    = (const float*)d_in[13];
    const float* b3_l    = (const float*)d_in[14];
    const float* w3_r    = (const float*)d_in[15];
    const float* bn1_g   = (const float*)d_in[16];
    const float* bn1_b   = (const float*)d_in[17];
    const float* bn1_m   = (const float*)d_in[18];
    const float* bn1_v   = (const float*)d_in[19];
    const float* bn2_g   = (const float*)d_in[20];
    const float* bn2_b   = (const float*)d_in[21];
    const float* bn2_m   = (const float*)d_in[22];
    const float* bn2_v   = (const float*)d_in[23];

    const int N = in_sizes[0] / FIN;        // 100000
    const int E = in_sizes[2] / 2;          // 1600000

    // ---- workspace carve-up (aligned to 256 B) ----
    char* ws = (char*)d_ws;
    size_t off = 0;
    auto carve = [&](size_t bytes) -> char* {
        char* p = ws + off;
        off += (bytes + 255) & ~(size_t)255;
        return p;
    };
    float*  EW   = (float*) carve((size_t)E * 4);            //  6.4 MB
    float*  RDEG = (float*) carve((size_t)N * 4);            //  0.4 MB
    float*  S    = (float*) carve((size_t)N * FIN * 4);      // 51.2 MB
    float*  H    = (float*) carve((size_t)N * FIN * 4);      // 51.2 MB
    __bf16* AGGB = (__bf16*)carve((size_t)N * FIN * 2);      // 25.6 MB
    __bf16* XB   = (__bf16*)carve((size_t)N * FIN * 2);      // 25.6 MB
    __bf16* P1L  = (__bf16*)carve(128 * 128 * 2);
    __bf16* P1R  = (__bf16*)carve(128 * 128 * 2);
    __bf16* P2L  = (__bf16*)carve(128 * 128 * 2);
    __bf16* P2R  = (__bf16*)carve(128 * 128 * 2);
    __bf16* P3L  = (__bf16*)carve(4 * 512 * 2);
    __bf16* P3R  = (__bf16*)carve(4 * 512 * 2);
    (void)ws_size;  // ~161 MB used

    const int TPB = 256;
    int eb   = (E + TPB - 1) / TPB;                 // edge-parallel blocks
    int nb   = (N + TPB - 1) / TPB;
    int scb  = (int)(((long long)E * 32 + TPB - 1) / TPB);
    int elb  = (N * FIN + TPB - 1) / TPB;           // elementwise over N*128
    int row_tiles = N / 16;                         // 6250 (N divisible by 16)

    // ---- prep: degree, edge weights, packed weights ----
    hipMemsetAsync(RDEG, 0, (size_t)N * 4, stream);
    deg_kernel<<<eb, TPB, 0, stream>>>(ei, RDEG, E);
    rdeg_kernel<<<nb, TPB, 0, stream>>>(RDEG, N);
    edge_mlp_kernel<<<eb, TPB, 0, stream>>>(ea, ew1_w, ew1_b, ew2_w, ew2_b, EW, E);

    int pw128 = (4 * 8 * 512 + TPB - 1) / TPB;      // 16384 elems
    int pw16  = (4 * 1 * 512 + TPB - 1) / TPB;      //  2048 elems
    pack_w_kernel<<<pw128, TPB, 0, stream>>>(w1_l, P1L, 128, 8);
    pack_w_kernel<<<pw128, TPB, 0, stream>>>(w1_r, P1R, 128, 8);
    pack_w_kernel<<<pw128, TPB, 0, stream>>>(w2_l, P2L, 128, 8);
    pack_w_kernel<<<pw128, TPB, 0, stream>>>(w2_r, P2R, 128, 8);
    pack_w_kernel<<<pw16,  TPB, 0, stream>>>(w3_l, P3L, 2, 1);
    pack_w_kernel<<<pw16,  TPB, 0, stream>>>(w3_r, P3R, 2, 1);

    // ---- layer 1: x -> H ----
    hipMemsetAsync(S, 0, (size_t)N * FIN * 4, stream);
    scatter_kernel<<<scb, TPB, 0, stream>>>(x, ei, EW, S, E);
    aggfin_kernel<<<elb, TPB, 0, stream>>>(S, RDEG, x, AGGB, XB, N * FIN);
    {
        int tiles = row_tiles * 8;
        int gb = (tiles * 32 + TPB - 1) / TPB;
        sage_gemm_kernel<<<gb, TPB, 0, stream>>>(AGGB, XB, P1L, P1R, b1_l,
                                                 bn1_g, bn1_b, bn1_m, bn1_v, 1,
                                                 H, 128, 128, 8, row_tiles);
    }

    // ---- layer 2: H -> H (in place; GEMM reads only bf16 copies) ----
    hipMemsetAsync(S, 0, (size_t)N * FIN * 4, stream);
    scatter_kernel<<<scb, TPB, 0, stream>>>(H, ei, EW, S, E);
    aggfin_kernel<<<elb, TPB, 0, stream>>>(S, RDEG, H, AGGB, XB, N * FIN);
    {
        int tiles = row_tiles * 8;
        int gb = (tiles * 32 + TPB - 1) / TPB;
        sage_gemm_kernel<<<gb, TPB, 0, stream>>>(AGGB, XB, P2L, P2R, b2_l,
                                                 bn2_g, bn2_b, bn2_m, bn2_v, 1,
                                                 H, 128, 128, 8, row_tiles);
    }

    // ---- layer 3: H -> d_out [N,2] ----
    hipMemsetAsync(S, 0, (size_t)N * FIN * 4, stream);
    scatter_kernel<<<scb, TPB, 0, stream>>>(H, ei, EW, S, E);
    aggfin_kernel<<<elb, TPB, 0, stream>>>(S, RDEG, H, AGGB, XB, N * FIN);
    {
        int tiles = row_tiles * 1;
        int gb = (tiles * 32 + TPB - 1) / TPB;
        sage_gemm_kernel<<<gb, TPB, 0, stream>>>(AGGB, XB, P3L, P3R, b3_l,
                                                 b3_l, b3_l, b3_l, b3_l, 0,
                                                 (float*)d_out, 2, 2, 1, row_tiles);
    }
}
